// sparc_25151328485432
// MI455X (gfx1250) — matrix operand order repaired and checked by reading, not compiled, not run on hardware
//
#include <hip/hip_runtime.h>
#include <stdint.h>

// ---------------------------------------------------------------------------
// SPARC alignment pipeline for MI455X (gfx1250, wave32).
//  * All GEMMs: v_wmma_f32_16x16x32_bf16, f32 accumulation, bf16 operands.
//  * Big visual GEMMs: A-activations staged through LDS by the Tensor Data
//    Mover (TENSOR_LOAD_TO_LDS, TENSORcnt, double buffered); B (L2-resident
//    weights) loaded per K-step as one clause.
//  * Small GEMMs: register software pipelining.
// ---------------------------------------------------------------------------

typedef __bf16 bf16;
typedef __attribute__((ext_vector_type(8)))  __bf16 v8bf;
typedef __attribute__((ext_vector_type(16))) __bf16 v16bf;
typedef __attribute__((ext_vector_type(8)))  float  v8f;
typedef __attribute__((ext_vector_type(4)))  unsigned int u32x4;
typedef __attribute__((ext_vector_type(8)))  int i32x8;
typedef __attribute__((ext_vector_type(4)))  int i32x4;

union Frag16 { v16bf v; v8bf h[2]; };

#define SIGMA (1.0f / 625.0f)

// ---- problem constants ----
#define BB   128
#define PP   625
#define TT   77
#define DV   768
#define DT   512
#define MV   (BB * PP)      // 80000 visual rows
#define MT   (BB * TT)      // 9856 text rows
#define PPAD 640            // P padded to multiple of 16

// ---------------------------------------------------------------------------
// wave reductions (wave32)
// ---------------------------------------------------------------------------
__device__ __forceinline__ float wred_sum(float v) {
#pragma unroll
  for (int m = 16; m >= 1; m >>= 1) v += __shfl_xor(v, m, 32);
  return v;
}
__device__ __forceinline__ float wred_min(float v) {
#pragma unroll
  for (int m = 16; m >= 1; m >>= 1) v = fminf(v, __shfl_xor(v, m, 32));
  return v;
}
__device__ __forceinline__ float wred_max(float v) {
#pragma unroll
  for (int m = 16; m >= 1; m >>= 1) v = fmaxf(v, __shfl_xor(v, m, 32));
  return v;
}

// ---------------------------------------------------------------------------
// WMMA fragment loaders (CDNA5 ISA 7.12.2, 16-bit, wave32)
// A 16x32 (MxK): lanes 0-15 row=lane, K chunks [0..7],[16..23]
//                lanes16-31 row=lane-16, K chunks [8..15],[24..31]
// ---------------------------------------------------------------------------
__device__ __forceinline__ v16bf load_a_frag(const bf16* __restrict__ A, int lda,
                                             int m0, int kbase, int lane, int mclamp) {
  int r = lane & 15, half = lane >> 4;
  int m = m0 + r; if (m > mclamp) m = mclamp;
  const bf16* p = A + (size_t)m * lda + kbase + half * 8;
  Frag16 f;
  f.h[0] = *(const v8bf*)(p);
  f.h[1] = *(const v8bf*)(p + 16);
  return f.v;
}

// A fragment out of an LDS-staged 128x32 tile (row stride 32 elements)
__device__ __forceinline__ v16bf load_a_frag_lds(const bf16* tile, int wave, int lane) {
  int r = lane & 15, half = lane >> 4;
  const bf16* p = tile + (wave * 16 + r) * 32 + half * 8;
  Frag16 f;
  f.h[0] = *(const v8bf*)(p);
  f.h[1] = *(const v8bf*)(p + 16);
  return f.v;
}

// B 32x16 (KxN) from N-major storage Bt[n][k]: lane=col, K chunks as A fragment
__device__ __forceinline__ v16bf load_b_frag(const bf16* __restrict__ Bt, int ldb,
                                             int n0, int kbase, int lane, int nclamp) {
  int n = lane & 15, half = lane >> 4;
  int c = n0 + n; if (c > nclamp) c = nclamp;
  const bf16* p = Bt + (size_t)c * ldb + kbase + half * 8;
  Frag16 f;
  f.h[0] = *(const v8bf*)(p);
  f.h[1] = *(const v8bf*)(p + 16);
  return f.v;
}

// ---------------------------------------------------------------------------
// TDM: async-load a 128-row x 32-col bf16 tile (row stride = ldg elements)
// into LDS at byte offset lds_off.  2-D descriptor per CDNA5 ISA ch.8.
// ---------------------------------------------------------------------------
__device__ __forceinline__ void tdm_load_tile_128x32(const bf16* gptr,
                                                     unsigned lds_off, int ldg) {
  unsigned long long ga = (unsigned long long)(uintptr_t)gptr;
  u32x4 g0;
  g0[0] = 1u;                                             // count=1, load mode
  g0[1] = lds_off;                                        // LDS byte address
  g0[2] = (unsigned)(ga & 0xFFFFFFFFu);                   // global_addr[31:0]
  g0[3] = (unsigned)((ga >> 32) & 0x1FFFFFFu) | (2u << 30); // addr[56:32]|type=2
  i32x8 g1;
  g1[0] = 0x00010000;                                     // data_size=1 (2 bytes)
  g1[1] = (int)(((unsigned)ldg & 0xFFFFu) << 16);         // tensor_dim0[15:0]
  g1[2] = (int)(((unsigned)ldg >> 16) | (128u << 16));    // dim0 hi | tensor_dim1=128
  g1[3] = (int)(32u << 16);                               // tile_dim0 = 32
  g1[4] = 128;                                            // tile_dim1 = 128
  g1[5] = ldg;                                            // tensor_dim0_stride lo
  g1[6] = 0;
  g1[7] = 0;
  i32x4 z4 = {0, 0, 0, 0};
#if __clang_major__ >= 23
  i32x8 z8 = {0, 0, 0, 0, 0, 0, 0, 0};
  __builtin_amdgcn_tensor_load_to_lds(g0, g1, z4, z4, z8, 0);
#else
  __builtin_amdgcn_tensor_load_to_lds(g0, g1, z4, z4, 0);
#endif
}

// ---------------------------------------------------------------------------
// weight transpose+convert: out[n*K + k] = (bf16) in[k*N + n]
// ---------------------------------------------------------------------------
__global__ void k_transpose_w(const float* __restrict__ in, bf16* __restrict__ out,
                              int K, int N) {
  int idx = blockIdx.x * 256 + threadIdx.x;
  if (idx >= K * N) return;
  int n = idx / K, k = idx % K;
  out[idx] = (bf16)in[(size_t)k * N + n];
}

// ---------------------------------------------------------------------------
// LayerNorm (wave per row) -> bf16
// ---------------------------------------------------------------------------
__global__ void k_ln_visual(const float* __restrict__ x, const float* __restrict__ g,
                            const float* __restrict__ b, bf16* __restrict__ out) {
  int lane = threadIdx.x & 31, wave = threadIdx.x >> 5;
  int row = blockIdx.x * 8 + wave;                 // 80000 rows
  const float* xr = x + (size_t)row * DV;
  float vals[24], s = 0.f, ss = 0.f;
#pragma unroll
  for (int j = 0; j < 24; ++j) {
    float v = xr[lane + j * 32];
    vals[j] = v; s += v; ss += v * v;
  }
  s = wred_sum(s); ss = wred_sum(ss);
  float mean = s * (1.f / DV);
  float var  = ss * (1.f / DV) - mean * mean;
  float inv  = rsqrtf(var + 1e-5f);
  bf16* orow = out + (size_t)row * DV;
#pragma unroll
  for (int j = 0; j < 24; ++j) {
    int idx = lane + j * 32;
    orow[idx] = (bf16)((vals[j] - mean) * inv * g[idx] + b[idx]);
  }
}

__global__ void k_ln_text(const float* __restrict__ x, const float* __restrict__ g,
                          const float* __restrict__ b, bf16* __restrict__ out) {
  int lane = threadIdx.x & 31, wave = threadIdx.x >> 5;
  int row = blockIdx.x * 8 + wave;                 // 9856 rows
  const float* xr = x + (size_t)row * DT;
  float vals[16], s = 0.f, ss = 0.f;
#pragma unroll
  for (int j = 0; j < 16; ++j) {
    float v = xr[lane + j * 32];
    vals[j] = v; s += v; ss += v * v;
  }
  s = wred_sum(s); ss = wred_sum(ss);
  float mean = s * (1.f / DT);
  float var  = ss * (1.f / DT) - mean * mean;
  float inv  = rsqrtf(var + 1e-5f);
  bf16* orow = out + (size_t)row * DT;
#pragma unroll
  for (int j = 0; j < 16; ++j) {
    int idx = lane + j * 32;
    orow[idx] = (bf16)((vals[j] - mean) * inv * g[idx] + b[idx]);
  }
}

// ---------------------------------------------------------------------------
// GEMM1 (TDM-staged): [80000 x 768] @ [768 x 1024]
// Block = 128 rows x 64 cols; wave w owns rows [w*16, w*16+16).
// A slices (128x32) double-buffered in LDS via TENSOR_LOAD_TO_LDS.
// cols 0..511 -> +bias -> f32 v_patch partial ; 512..1023 -> +bias -> GELU bf16
// ---------------------------------------------------------------------------
__global__ void __launch_bounds__(256, 1)
k_gemm_visual1(const bf16* __restrict__ vp, const bf16* __restrict__ Wcat,
               const float* __restrict__ lin_b, const float* __restrict__ nl_b1,
               float* __restrict__ outp, bf16* __restrict__ hidden) {
  __shared__ bf16 smem[2][128 * 32];               // 2 x 8 KB double buffer
  int lane = threadIdx.x & 31, wave = threadIdx.x >> 5;
  int mblk = blockIdx.x >> 4;                      // 625 row-blocks
  int ng   = blockIdx.x & 15;                      // 16 col-groups of 64
  int m0   = mblk * 128;
  int n0   = ng * 64;
  const bf16* Ag = vp + (size_t)m0 * DV;

  if (wave == 0) tdm_load_tile_128x32(Ag, 0u, DV);

  v8f acc[4] = {};
  const int NK = DV / 32;                          // 24 K-steps
  for (int kc = 0; kc < NK; ++kc) {
    if (wave == 0) {
      if (kc + 1 < NK) {
        tdm_load_tile_128x32(Ag + (kc + 1) * 32, (unsigned)(((kc + 1) & 1) * 8192), DV);
        __builtin_amdgcn_s_wait_tensorcnt((short)1);   // current slice complete
      } else {
        __builtin_amdgcn_s_wait_tensorcnt((short)0);
      }
    }
    __syncthreads();                               // publish slice to all waves
    v16bf a = load_a_frag_lds(&smem[kc & 1][0], wave, lane);
    v16bf bf0 = load_b_frag(Wcat, DV, n0,      kc * 32, lane, 1023);
    v16bf bf1 = load_b_frag(Wcat, DV, n0 + 16, kc * 32, lane, 1023);
    v16bf bf2 = load_b_frag(Wcat, DV, n0 + 32, kc * 32, lane, 1023);
    v16bf bf3 = load_b_frag(Wcat, DV, n0 + 48, kc * 32, lane, 1023);
    acc[0] = __builtin_amdgcn_wmma_f32_16x16x32_bf16(false, a, false, bf0, (short)0, acc[0], false, false);
    acc[1] = __builtin_amdgcn_wmma_f32_16x16x32_bf16(false, a, false, bf1, (short)0, acc[1], false, false);
    acc[2] = __builtin_amdgcn_wmma_f32_16x16x32_bf16(false, a, false, bf2, (short)0, acc[2], false, false);
    acc[3] = __builtin_amdgcn_wmma_f32_16x16x32_bf16(false, a, false, bf3, (short)0, acc[3], false, false);
    __syncthreads();                               // retire slice before reuse
  }

  int mw0 = m0 + wave * 16;
  int n = lane & 15, moff = (lane >> 4) * 8;
#pragma unroll
  for (int j = 0; j < 4; ++j) {
    int col = n0 + j * 16 + n;
#pragma unroll
    for (int r = 0; r < 8; ++r) {
      int row = mw0 + moff + r;
      float v = acc[j][r];
      if (col < DT) {
        outp[(size_t)row * DT + col] = v + lin_b[col];
      } else {
        int c2 = col - DT;
        float xx = v + nl_b1[c2];
        float ge = 0.5f * xx * (1.0f + erff(xx * 0.70710678118654752f));
        hidden[(size_t)row * DT + c2] = (bf16)ge;
      }
    }
  }
}

// GEMM2 (TDM-staged): v_patch = partial + hidden @ W2 + b2
__global__ void __launch_bounds__(256, 1)
k_gemm_visual2(const bf16* __restrict__ hidden, const bf16* __restrict__ W2t,
               const float* __restrict__ b2, float* __restrict__ outp,
               bf16* __restrict__ vpe_bf) {
  __shared__ bf16 smem[2][128 * 32];
  int lane = threadIdx.x & 31, wave = threadIdx.x >> 5;
  int mblk = blockIdx.x >> 3;                      // 625 row-blocks
  int ng   = blockIdx.x & 7;                       // 8 col-groups of 64
  int m0   = mblk * 128;
  int n0   = ng * 64;
  const bf16* Ag = hidden + (size_t)m0 * DT;

  if (wave == 0) tdm_load_tile_128x32(Ag, 0u, DT);

  v8f acc[4] = {};
  const int NK = DT / 32;                          // 16 K-steps
  for (int kc = 0; kc < NK; ++kc) {
    if (wave == 0) {
      if (kc + 1 < NK) {
        tdm_load_tile_128x32(Ag + (kc + 1) * 32, (unsigned)(((kc + 1) & 1) * 8192), DT);
        __builtin_amdgcn_s_wait_tensorcnt((short)1);
      } else {
        __builtin_amdgcn_s_wait_tensorcnt((short)0);
      }
    }
    __syncthreads();
    v16bf a = load_a_frag_lds(&smem[kc & 1][0], wave, lane);
    v16bf bf0 = load_b_frag(W2t, DT, n0,      kc * 32, lane, DT - 1);
    v16bf bf1 = load_b_frag(W2t, DT, n0 + 16, kc * 32, lane, DT - 1);
    v16bf bf2 = load_b_frag(W2t, DT, n0 + 32, kc * 32, lane, DT - 1);
    v16bf bf3 = load_b_frag(W2t, DT, n0 + 48, kc * 32, lane, DT - 1);
    acc[0] = __builtin_amdgcn_wmma_f32_16x16x32_bf16(false, a, false, bf0, (short)0, acc[0], false, false);
    acc[1] = __builtin_amdgcn_wmma_f32_16x16x32_bf16(false, a, false, bf1, (short)0, acc[1], false, false);
    acc[2] = __builtin_amdgcn_wmma_f32_16x16x32_bf16(false, a, false, bf2, (short)0, acc[2], false, false);
    acc[3] = __builtin_amdgcn_wmma_f32_16x16x32_bf16(false, a, false, bf3, (short)0, acc[3], false, false);
    __syncthreads();
  }

  int mw0 = m0 + wave * 16;
  int n = lane & 15, moff = (lane >> 4) * 8;
#pragma unroll
  for (int j = 0; j < 4; ++j) {
    int col = n0 + j * 16 + n;
#pragma unroll
    for (int r = 0; r < 8; ++r) {
      int row = mw0 + moff + r;
      size_t o = (size_t)row * DT + col;
      float v = outp[o] + acc[j][r] + b2[col];
      outp[o] = v;
      vpe_bf[o] = (bf16)v;
    }
  }
}

// GEMM text (register-pipelined): lte = ln_text @ Tw + b  [9856 x 512], K=512
__global__ void __launch_bounds__(256, 1)
k_gemm_text(const bf16* __restrict__ lntx, const bf16* __restrict__ Tt,
            const float* __restrict__ tb, float* __restrict__ lte) {
  int lane = threadIdx.x & 31, wave = threadIdx.x >> 5;
  int wid = blockIdx.x * 8 + wave;                 // 616 mtiles * 8 ngroups
  int mt = wid >> 3, ng = wid & 7;
  int m0 = mt * 16, n0 = ng * 64;
  v8f acc[4] = {};
  v16bf a = load_a_frag(lntx, DT, m0, 0, lane, MT - 1);
  v16bf bcur[4];
#pragma unroll
  for (int j = 0; j < 4; ++j) bcur[j] = load_b_frag(Tt, DT, n0 + j * 16, 0, lane, DT - 1);
  for (int k = 32; k < DT; k += 32) {
    v16bf an = load_a_frag(lntx, DT, m0, k, lane, MT - 1);
    v16bf bn[4];
#pragma unroll
    for (int j = 0; j < 4; ++j) bn[j] = load_b_frag(Tt, DT, n0 + j * 16, k, lane, DT - 1);
#pragma unroll
    for (int j = 0; j < 4; ++j)
      acc[j] = __builtin_amdgcn_wmma_f32_16x16x32_bf16(
          false, a, false, bcur[j], (short)0, acc[j], false, false);
    a = an;
#pragma unroll
    for (int j = 0; j < 4; ++j) bcur[j] = bn[j];
  }
#pragma unroll
  for (int j = 0; j < 4; ++j)
    acc[j] = __builtin_amdgcn_wmma_f32_16x16x32_bf16(
        false, a, false, bcur[j], (short)0, acc[j], false, false);

  int n = lane & 15, moff = (lane >> 4) * 8;
#pragma unroll
  for (int j = 0; j < 4; ++j) {
    int col = n0 + j * 16 + n;
#pragma unroll
    for (int r = 0; r < 8; ++r)
      lte[(size_t)(m0 + moff + r) * DT + col] = acc[j][r] + tb[col];
  }
}

// token L2-normalize -> d_out region1 ; bf16 copy of UNnormalized lte for sim
__global__ void k_token_norm(const float* __restrict__ lte, float* __restrict__ outn,
                             bf16* __restrict__ lte_bf) {
  int lane = threadIdx.x & 31, wave = threadIdx.x >> 5;
  int row = blockIdx.x * 8 + wave;                 // 9856 rows
  const float* xr = lte + (size_t)row * DT;
  float vals[16], ss = 0.f;
#pragma unroll
  for (int j = 0; j < 16; ++j) { float v = xr[lane + j * 32]; vals[j] = v; ss += v * v; }
  ss = wred_sum(ss);
  float inv = 1.f / fmaxf(sqrtf(ss), 1e-12f);
#pragma unroll
  for (int j = 0; j < 16; ++j) {
    int idx = lane + j * 32;
    outn[(size_t)row * DT + idx]   = vals[j] * inv;
    lte_bf[(size_t)row * DT + idx] = (bf16)vals[j];
  }
}

// transpose vpe [b][p][d] -> vpeT [b][d][p(pad 640)]
__global__ void k_transpose_vpe(const bf16* __restrict__ vpe, bf16* __restrict__ vpeT) {
  int idx = blockIdx.x * 256 + threadIdx.x;        // 128*512*640
  int p = idx % PPAD; int t = idx / PPAD; int d = t % DT; int b = t / DT;
  bf16 v = (p < PP) ? vpe[((size_t)b * PP + p) * DT + d] : (bf16)0.f;
  vpeT[idx] = v;
}

// sim = lte @ vpe^T per batch: M=77(5 tiles), N=625(pad 640), K=512
__global__ void __launch_bounds__(256, 1)
k_gemm_sim(const bf16* __restrict__ lte_bf, const bf16* __restrict__ vpe_bf,
           float* __restrict__ sim) {
  int lane = threadIdx.x & 31, wave = threadIdx.x >> 5;
  int wid = blockIdx.x * 8 + wave;                 // 128 * 5 * 10
  int b = wid / 50; int rem = wid % 50;
  int mt = rem / 10, ng = rem % 10;
  int m0 = mt * 16, n0 = ng * 64;
  const bf16* A  = lte_bf + (size_t)b * TT * DT;
  const bf16* Bt = vpe_bf + (size_t)b * PP * DT;   // row p contiguous in d(=K)
  v8f acc[4] = {};
  v16bf a = load_a_frag(A, DT, m0, 0, lane, TT - 1);
  v16bf bcur[4];
#pragma unroll
  for (int j = 0; j < 4; ++j) bcur[j] = load_b_frag(Bt, DT, n0 + j * 16, 0, lane, PP - 1);
  for (int k = 32; k < DT; k += 32) {
    v16bf an = load_a_frag(A, DT, m0, k, lane, TT - 1);
    v16bf bn[4];
#pragma unroll
    for (int j = 0; j < 4; ++j) bn[j] = load_b_frag(Bt, DT, n0 + j * 16, k, lane, PP - 1);
#pragma unroll
    for (int j = 0; j < 4; ++j)
      acc[j] = __builtin_amdgcn_wmma_f32_16x16x32_bf16(
          false, a, false, bcur[j], (short)0, acc[j], false, false);
    a = an;
#pragma unroll
    for (int j = 0; j < 4; ++j) bcur[j] = bn[j];
  }
#pragma unroll
  for (int j = 0; j < 4; ++j)
    acc[j] = __builtin_amdgcn_wmma_f32_16x16x32_bf16(
        false, a, false, bcur[j], (short)0, acc[j], false, false);

  int n = lane & 15, moff = (lane >> 4) * 8;
#pragma unroll
  for (int j = 0; j < 4; ++j) {
    int p = n0 + j * 16 + n;
#pragma unroll
    for (int r = 0; r < 8; ++r) {
      int t = m0 + moff + r;
      if (t < TT && p < PP)
        sim[((size_t)b * TT + t) * PPAD + p] = acc[j][r];
    }
  }
}

// per (b,t): min/max normalize, sigma-threshold, sum-normalize -> bf16 weights
__global__ void k_sim_weights(const float* __restrict__ sim, bf16* __restrict__ wgt) {
  int lane = threadIdx.x & 31, wave = threadIdx.x >> 5;
  int row = blockIdx.x * 8 + wave;                 // 9856 rows
  const float* sr = sim + (size_t)row * PPAD;
  float v[20], mn = 3.4e38f, mx = -3.4e38f;
#pragma unroll
  for (int i = 0; i < 20; ++i) {
    int p = i * 32 + lane;
    if (p < PP) { float x = sr[p]; v[i] = x; mn = fminf(mn, x); mx = fmaxf(mx, x); }
    else v[i] = 0.f;
  }
  mn = wred_min(mn); mx = wred_max(mx);
  float scale = 1.f / (mx - mn + 1e-8f);
  float s = 0.f;
#pragma unroll
  for (int i = 0; i < 20; ++i) {
    int p = i * 32 + lane;
    if (p < PP) {
      float x = (v[i] - mn) * scale;
      x = (x < SIGMA) ? 0.f : x;
      v[i] = x; s += x;
    }
  }
  s = wred_sum(s);
  float inv = 1.f / (s + 1e-8f);
  bf16* wr = wgt + (size_t)row * PPAD;
#pragma unroll
  for (int i = 0; i < 20; ++i) {
    int p = i * 32 + lane;                         // p < 640 always
    wr[p] = (p < PP) ? (bf16)(v[i] * inv) : (bf16)0.f;
  }
}

// l_grouped_raw = w @ vpe per batch: M=77(5 tiles), N=512, K=640(20 steps)
__global__ void __launch_bounds__(256, 1)
k_gemm_grouped(const bf16* __restrict__ wgt, const bf16* __restrict__ vpeT,
               float* __restrict__ lgr) {
  int lane = threadIdx.x & 31, wave = threadIdx.x >> 5;
  int wid = blockIdx.x * 8 + wave;                 // 128 * 5 * 8
  int b = wid / 40; int rem = wid % 40;
  int mt = rem / 8, ng = rem % 8;
  int m0 = mt * 16, n0 = ng * 64;
  const bf16* A  = wgt  + (size_t)b * TT * PPAD;   // row t contiguous in p(=K)
  const bf16* Bt = vpeT + (size_t)b * DT * PPAD;   // row d contiguous in p(=K)
  v8f acc[4] = {};
  v16bf a = load_a_frag(A, PPAD, m0, 0, lane, TT - 1);
  v16bf bcur[4];
#pragma unroll
  for (int j = 0; j < 4; ++j) bcur[j] = load_b_frag(Bt, PPAD, n0 + j * 16, 0, lane, DT - 1);
  for (int k = 32; k < PPAD; k += 32) {
    v16bf an = load_a_frag(A, PPAD, m0, k, lane, TT - 1);
    v16bf bn[4];
#pragma unroll
    for (int j = 0; j < 4; ++j) bn[j] = load_b_frag(Bt, PPAD, n0 + j * 16, k, lane, DT - 1);
#pragma unroll
    for (int j = 0; j < 4; ++j)
      acc[j] = __builtin_amdgcn_wmma_f32_16x16x32_bf16(
          false, a, false, bcur[j], (short)0, acc[j], false, false);
    a = an;
#pragma unroll
    for (int j = 0; j < 4; ++j) bcur[j] = bn[j];
  }
#pragma unroll
  for (int j = 0; j < 4; ++j)
    acc[j] = __builtin_amdgcn_wmma_f32_16x16x32_bf16(
        false, a, false, bcur[j], (short)0, acc[j], false, false);

  int n = lane & 15, moff = (lane >> 4) * 8;
#pragma unroll
  for (int j = 0; j < 4; ++j) {
    int col = n0 + j * 16 + n;
#pragma unroll
    for (int r = 0; r < 8; ++r) {
      int t = m0 + moff + r;
      if (t < TT)
        lgr[((size_t)b * TT + t) * DT + col] = acc[j][r];
    }
  }
}

// L2-normalize rows [9856 x 512] -> d_out region2
__global__ void k_norm_rows(const float* __restrict__ x, float* __restrict__ out) {
  int lane = threadIdx.x & 31, wave = threadIdx.x >> 5;
  int row = blockIdx.x * 8 + wave;
  const float* xr = x + (size_t)row * DT;
  float vals[16], ss = 0.f;
#pragma unroll
  for (int j = 0; j < 16; ++j) { float v = xr[lane + j * 32]; vals[j] = v; ss += v * v; }
  ss = wred_sum(ss);
  float inv = 1.f / fmaxf(sqrtf(ss), 1e-12f);
#pragma unroll
  for (int j = 0; j < 16; ++j)
    out[(size_t)row * DT + lane + j * 32] = vals[j] * inv;
}

__global__ void k_copy_mask(const float* __restrict__ in, float* __restrict__ out, int n) {
  int i = blockIdx.x * 256 + threadIdx.x;
  if (i < n) out[i] = in[i];
}

// ---------------------------------------------------------------------------
extern "C" void kernel_launch(void* const* d_in, const int* in_sizes, int n_in,
                              void* d_out, int out_size, void* d_ws, size_t ws_size,
                              hipStream_t stream) {
  (void)in_sizes; (void)n_in; (void)out_size; (void)ws_size;
  const float* visual   = (const float*)d_in[0];
  const float* text     = (const float*)d_in[1];
  const float* mask     = (const float*)d_in[2];
  const float* v_ln_g   = (const float*)d_in[3];
  const float* v_ln_b   = (const float*)d_in[4];
  const float* v_lin_w  = (const float*)d_in[5];
  const float* v_lin_b  = (const float*)d_in[6];
  const float* v_nl_w1  = (const float*)d_in[7];
  const float* v_nl_b1  = (const float*)d_in[8];
  const float* v_nl_w2  = (const float*)d_in[9];
  const float* v_nl_b2  = (const float*)d_in[10];
  const float* t_ln_g   = (const float*)d_in[11];
  const float* t_ln_b   = (const float*)d_in[12];
  const float* t_lin_w  = (const float*)d_in[13];
  const float* t_lin_b  = (const float*)d_in[14];

  // d_out regions (f32): v_patch_embed | l_token_norm | l_grouped | mask
  float* out_vpe  = (float*)d_out;
  float* out_tok  = out_vpe + (size_t)MV * DT;
  float* out_grp  = out_tok + (size_t)MT * DT;
  float* out_mask = out_grp + (size_t)MT * DT;

  // workspace carve-up (all sub-buffer sizes are multiples of 256 B)
  char* w = (char*)d_ws;
  bf16*  WCAT   = (bf16*)w;  w += (size_t)1024 * DV * 2;        // [1024][768]
  bf16*  W2T    = (bf16*)w;  w += (size_t)DT * DT * 2;          // [512][512]
  bf16*  TTW    = (bf16*)w;  w += (size_t)DT * DT * 2;          // [512][512]
  bf16*  VP     = (bf16*)w;  w += (size_t)MV * DV * 2;          // ln visual bf16
  bf16*  HID    = (bf16*)w;  w += (size_t)MV * DT * 2;          // gelu hidden bf16
  bf16*  VPEB   = (bf16*)w;  w += (size_t)MV * DT * 2;          // v_patch bf16
  bf16*  VPET   = (bf16*)w;  w += (size_t)BB * DT * PPAD * 2;   // transposed, padded
  bf16*  LNTX   = (bf16*)w;  w += (size_t)MT * DT * 2;          // ln text bf16
  float* LTE    = (float*)w; w += (size_t)MT * DT * 4;          // l_token_embed f32
  bf16*  LTEB   = (bf16*)w;  w += (size_t)MT * DT * 2;          // l_token_embed bf16
  float* SIM    = (float*)w; w += (size_t)MT * PPAD * 4;        // sim f32 (pad 640)
  bf16*  WGT    = (bf16*)w;  w += (size_t)MT * PPAD * 2;        // weights bf16
  float* LGR    = (float*)w; w += (size_t)MT * DT * 4;          // l_grouped raw f32

  dim3 blk(256);

  // 1) weights -> bf16, N-major (transposed)
  k_transpose_w<<<dim3((DV * DT + 255) / 256), blk, 0, stream>>>(v_lin_w, WCAT, DV, DT);
  k_transpose_w<<<dim3((DV * DT + 255) / 256), blk, 0, stream>>>(v_nl_w1, WCAT + (size_t)DT * DV, DV, DT);
  k_transpose_w<<<dim3((DT * DT + 255) / 256), blk, 0, stream>>>(v_nl_w2, W2T, DT, DT);
  k_transpose_w<<<dim3((DT * DT + 255) / 256), blk, 0, stream>>>(t_lin_w, TTW, DT, DT);

  // 2) LayerNorms
  k_ln_visual<<<dim3(MV / 8), blk, 0, stream>>>(visual, v_ln_g, v_ln_b, VP);
  k_ln_text<<<dim3(MT / 8), blk, 0, stream>>>(text, t_ln_g, t_ln_b, LNTX);

  // 3) big GEMMs (TDM-staged A, 128-row blocks)
  k_gemm_visual1<<<dim3((MV / 128) * 16), blk, 0, stream>>>(VP, WCAT, v_lin_b, v_nl_b1, out_vpe, HID);
  k_gemm_visual2<<<dim3((MV / 128) * 8), blk, 0, stream>>>(HID, W2T, v_nl_b2, out_vpe, VPEB);
  k_gemm_text<<<dim3((MT / 16) * 8 / 8), blk, 0, stream>>>(LNTX, TTW, t_lin_b, LTE);

  // 4) token norm + transpose for P-contraction
  k_token_norm<<<dim3(MT / 8), blk, 0, stream>>>(LTE, out_tok, LTEB);
  k_transpose_vpe<<<dim3(BB * DT * PPAD / 256), blk, 0, stream>>>(VPEB, VPET);

  // 5) sim GEMM, row weights, grouped GEMM, final norms
  k_gemm_sim<<<dim3(BB * 5 * 10 / 8), blk, 0, stream>>>(LTEB, VPEB, SIM);
  k_sim_weights<<<dim3(MT / 8), blk, 0, stream>>>(SIM, WGT);
  k_gemm_grouped<<<dim3(BB * 5 * 8 / 8), blk, 0, stream>>>(WGT, VPET, LGR);
  k_norm_rows<<<dim3(MT / 8), blk, 0, stream>>>(LGR, out_grp);
  k_copy_mask<<<dim3((MT + 255) / 256), blk, 0, stream>>>(mask, out_mask, MT);
}
